// PointTransformerLayer_18339510354755
// MI455X (gfx1250) — matrix operand order repaired and checked by reading, not compiled, not run on hardware
//
#include <hip/hip_runtime.h>

// ---------------------------------------------------------------------------
// Point Transformer layer for MI455X (gfx1250, wave32, WMMA).
// B=2, N=8192, C=128, K=16.
// ---------------------------------------------------------------------------

typedef __attribute__((ext_vector_type(16))) __bf16 v16bf;
typedef __attribute__((ext_vector_type(8)))  __bf16 v8bf;
typedef __attribute__((ext_vector_type(8)))  float  v8f;

#define BB 2
#define NN 8192
#define CC 128
#define KK 16
#define QT 8            // queries per workgroup
#define RR (QT * KK)    // 128 rows (query,neighbor) per workgroup

// ---------------------------------------------------------------------------
// B-operand loader: packed weights hold, per (colTile ct, kTile kt, lane),
// 16 contiguous bf16 = column (ct*16 + lane&15), K = kt*32 + 16*(lane>>4)..+15
// (contiguous k order, same as the A loader: lanes 0-15 K=0..15, lanes 16-31 K=16..31)
// ---------------------------------------------------------------------------
__device__ __forceinline__ void load_Btile(const __bf16* __restrict__ Wp, int ct,
                                           int lane, v16bf* Breg) {
#pragma unroll
  for (int kt = 0; kt < 4; ++kt)
    Breg[kt] = *(const v16bf*)(Wp + ((size_t)((ct * 4 + kt) * 32 + lane)) * 16);
}

// ---------------------------------------------------------------------------
// One 16x128 row-tile x 128x16 col-tile GEMM via 4x v_wmma_f32_16x16x32_bf16.
// A is row-major bf16 [rows][128] in LDS. In the contiguous k order shared with B, lane l
// (m=l&15, hi=l>>4) needs row m, K chunks [32k+16hi, +8) and [32k+16hi+8, +8):
// two 16-byte ds_load_b128 per k-step.
// ---------------------------------------------------------------------------
__device__ __forceinline__ v8f wmma_rowtile(const __bf16* Abuf, int rowBase,
                                            const v16bf* Breg, int lane) {
  v8f acc = {};
  const int m = lane & 15;
  const int hi = lane >> 4;
  const __bf16* arow = Abuf + (rowBase + m) * CC;
#pragma unroll
  for (int kk = 0; kk < 4; ++kk) {
    v8bf c0 = *(const v8bf*)(arow + kk * 32 + 16 * hi);
    v8bf c1 = *(const v8bf*)(arow + kk * 32 + 16 * hi + 8);
    v16bf a;
#pragma unroll
    for (int i = 0; i < 8; ++i) { a[i] = c0[i]; a[i + 8] = c1[i]; }
    acc = __builtin_amdgcn_wmma_f32_16x16x32_bf16(false, a, false, Breg[kk],
                                                  (short)0, acc, false, false);
  }
  return acc;
}

// ---------------------------------------------------------------------------
// Kernel 1: brute-force kNN (top-16 by squared distance, includes self).
// One thread per query point; positions tiled through LDS.
// ---------------------------------------------------------------------------
__global__ __launch_bounds__(256) void knn_kernel(const float* __restrict__ pos,
                                                  int* __restrict__ outIdx) {
  __shared__ float spx[256], spy[256], spz[256];
  const int b = blockIdx.x / (NN / 256);
  const int n = (blockIdx.x % (NN / 256)) * 256 + threadIdx.x;
  const float* pq = pos + ((size_t)b * NN + n) * 3;
  const float qx = pq[0], qy = pq[1], qz = pq[2];
  float bd[KK];
  int bi[KK];
#pragma unroll
  for (int t = 0; t < KK; ++t) { bd[t] = 3.4e38f; bi[t] = 0; }

  for (int tile = 0; tile < NN; tile += 256) {
    const float* pt = pos + ((size_t)b * NN + tile + threadIdx.x) * 3;
    spx[threadIdx.x] = pt[0];
    spy[threadIdx.x] = pt[1];
    spz[threadIdx.x] = pt[2];
    __syncthreads();
    for (int j = 0; j < 256; ++j) {
      const float dx = qx - spx[j], dy = qy - spy[j], dz = qz - spz[j];
      const float d2 = dx * dx + dy * dy + dz * dz;
      if (d2 < bd[KK - 1]) {
        float cd = d2;
        int ci = tile + j;
#pragma unroll
        for (int t = 0; t < KK; ++t) {
          if (cd < bd[t]) {
            const float td = bd[t]; const int ti = bi[t];
            bd[t] = cd; bi[t] = ci; cd = td; ci = ti;
          }
        }
      }
    }
    __syncthreads();
  }
  int* dst = outIdx + ((size_t)b * NN + n) * KK;
#pragma unroll
  for (int t = 0; t < KK; ++t) dst[t] = bi[t];
}

// ---------------------------------------------------------------------------
// Kernel 2: pre-swizzle a 128x128 fp32 weight (row-major, [k][n]) into the
// per-lane WMMA B-operand bf16 layout. grid = 32 (8 colTiles x 4 kTiles),
// block = 32 lanes; each lane writes its 16 contiguous bf16.
// ---------------------------------------------------------------------------
__global__ void pack_weight(const float* __restrict__ W, __bf16* __restrict__ P) {
  const int ct = blockIdx.x >> 2;
  const int kt = blockIdx.x & 3;
  const int lane = threadIdx.x;
  const int ncol = ct * 16 + (lane & 15);
  const int k0 = kt * 32 + 16 * (lane >> 4);
  __bf16* dst = P + ((size_t)blockIdx.x * 32 + lane) * 16;
#pragma unroll
  for (int i = 0; i < 16; ++i)
    dst[i] = (__bf16)W[(size_t)(k0 + i) * CC + ncol];
}

// ---------------------------------------------------------------------------
// Kernel 3: fused layer. One workgroup (8 waves / 256 threads) handles
// 8 queries x 16 neighbors = 128 rows through LDS-resident GEMM phases.
// Dynamic LDS carve (~140 KB; gfx1250 WGP has 320 KB).
// ---------------------------------------------------------------------------
__global__ __launch_bounds__(256) void pt_fused(
    const float* __restrict__ x, const float* __restrict__ pos,
    const int* __restrict__ knn,
    const __bf16* __restrict__ WqP, const __bf16* __restrict__ WkP,
    const __bf16* __restrict__ WvP, const __bf16* __restrict__ Wp2P,
    const __bf16* __restrict__ Wg1P, const __bf16* __restrict__ Wg2P,
    const float* __restrict__ bq, const float* __restrict__ bk,
    const float* __restrict__ bv, const float* __restrict__ Wp1,
    const float* __restrict__ bp1, const float* __restrict__ bp2,
    const float* __restrict__ bg1, const float* __restrict__ bg2,
    float* __restrict__ out) {
  extern __shared__ char smem[];
  __bf16* Abf  = (__bf16*)smem;         // [RR][CC] neighbor feats -> later g1 hidden
  __bf16* PEbf = Abf  + RR * CC;        // [RR][CC] positional encoding
  __bf16* Vbf  = PEbf + RR * CC;        // [RR][CC] v
  __bf16* Ubf  = Vbf  + RR * CC;        // [RR][CC] pe1 staging -> later u
  __bf16* qAbf = Ubf  + RR * CC;        // [16][CC] query rows (8 valid, 8 zero)
  float*  qRow = (float*)(qAbf + 16 * CC); // [QT][CC]
  float*  relp = qRow + QT * CC;        // [RR][4]
  float*  wp1s = relp + RR * 4;         // [3][CC]
  float*  bp1s = wp1s + 3 * CC;         // [CC]

  const int tid  = threadIdx.x;
  const int lane = tid & 31;
  const int wv   = tid >> 5;
  const int gq   = blockIdx.x * QT;     // flattened b*N + n
  const int b    = gq / NN;
  const int n0   = gq % NN;
  const int m    = lane & 15;
  const int hi   = lane >> 4;
  const float scl = 0.0883883476483184f; // 1/sqrt(128)

  // ---- stage Wp1/bp1 into LDS ----
  for (int i = tid; i < 3 * CC; i += 256) wp1s[i] = Wp1[i];
  if (tid < CC) bp1s[tid] = bp1[tid];

  // ---- gather neighbor features (fp32 -> bf16) + relative positions ----
  {
    const int r = tid >> 1, h = tid & 1;         // 2 threads per row
    const int j = knn[(size_t)gq * KK + r];
    const float4* src = (const float4*)(x + ((size_t)b * NN + j) * CC) + h * 16;
#pragma unroll
    for (int t = 0; t < 16; ++t) {
      const float4 f = src[t];
      const int c = h * 64 + t * 4;
      Abf[r * CC + c + 0] = (__bf16)f.x;
      Abf[r * CC + c + 1] = (__bf16)f.y;
      Abf[r * CC + c + 2] = (__bf16)f.z;
      Abf[r * CC + c + 3] = (__bf16)f.w;
    }
    if (h == 0) {
      const int qi = r >> 4;
      const float* pq = pos + ((size_t)b * NN + n0 + qi) * 3;
      const float* pj = pos + ((size_t)b * NN + j) * 3;
      relp[r * 4 + 0] = pq[0] - pj[0];
      relp[r * 4 + 1] = pq[1] - pj[1];
      relp[r * 4 + 2] = pq[2] - pj[2];
    }
  }
  // ---- stage query rows as bf16 A operand (rows QT..15 zeroed) ----
  if (tid < 64) {
    const int r = tid >> 2, ch = tid & 3;        // 16 rows x 4 chunks of 32
    if (r < QT) {
      const float4* src = (const float4*)(x + ((size_t)b * NN + n0 + r) * CC) + ch * 8;
#pragma unroll
      for (int t = 0; t < 8; ++t) {
        const float4 f = src[t];
        const int c = ch * 32 + t * 4;
        qAbf[r * CC + c + 0] = (__bf16)f.x;
        qAbf[r * CC + c + 1] = (__bf16)f.y;
        qAbf[r * CC + c + 2] = (__bf16)f.z;
        qAbf[r * CC + c + 3] = (__bf16)f.w;
      }
    } else {
      for (int c = ch * 32; c < ch * 32 + 32; ++c) qAbf[r * CC + c] = (__bf16)0.0f;
    }
  }
  __syncthreads();

  // ---- pe layer 1 (VALU): relu(rel @ Wp1 + bp1) -> Ubf ----
  {
    const int r = tid >> 1, h = tid & 1;
    const float rx = relp[r * 4 + 0], ry = relp[r * 4 + 1], rz = relp[r * 4 + 2];
#pragma unroll 8
    for (int c = h * 64; c < h * 64 + 64; ++c) {
      const float v = fmaf(rx, wp1s[c],
                      fmaf(ry, wp1s[CC + c],
                      fmaf(rz, wp1s[2 * CC + c], bp1s[c])));
      Ubf[r * CC + c] = (__bf16)fmaxf(v, 0.0f);
    }
  }
  // ---- q = x_q @ Wq + bq (one row tile; wave wv owns col tile wv) ----
  {
    v16bf Breg[4];
    load_Btile(WqP, wv, lane, Breg);
    v8f acc = wmma_rowtile(qAbf, 0, Breg, lane);
    const int col = wv * 16 + m;
    const float bias = bq[col];
#pragma unroll
    for (int vg = 0; vg < 8; ++vg) {
      const int row = vg + 8 * hi;
      if (row < QT) qRow[row * CC + col] = acc[vg] + bias;
    }
  }
  __syncthreads();

  // ---- pe layer 2: PE = pe1 @ Wp2 + bp2 -> PEbf ----
  {
    v16bf Breg[4];
    load_Btile(Wp2P, wv, lane, Breg);
    const int col = wv * 16 + m;
    const float bias = bp2[col];
    for (int rt = 0; rt < RR / 16; ++rt) {
      v8f acc = wmma_rowtile(Ubf, rt * 16, Breg, lane);
#pragma unroll
      for (int vg = 0; vg < 8; ++vg) {
        const int row = rt * 16 + vg + 8 * hi;
        PEbf[row * CC + col] = (__bf16)(acc[vg] + bias);
      }
    }
  }
  __syncthreads();

  // ---- v = nb @ Wv + bv -> Vbf ----
  {
    v16bf Breg[4];
    load_Btile(WvP, wv, lane, Breg);
    const int col = wv * 16 + m;
    const float bias = bv[col];
    for (int rt = 0; rt < RR / 16; ++rt) {
      v8f acc = wmma_rowtile(Abf, rt * 16, Breg, lane);
#pragma unroll
      for (int vg = 0; vg < 8; ++vg) {
        const int row = rt * 16 + vg + 8 * hi;
        Vbf[row * CC + col] = (__bf16)(acc[vg] + bias);
      }
    }
  }
  // ---- k = nb @ Wk + bk; u = (q - k + pe) * scale -> Ubf ----
  {
    v16bf Breg[4];
    load_Btile(WkP, wv, lane, Breg);
    const int col = wv * 16 + m;
    const float bias = bk[col];
    for (int rt = 0; rt < RR / 16; ++rt) {
      v8f acc = wmma_rowtile(Abf, rt * 16, Breg, lane);
#pragma unroll
      for (int vg = 0; vg < 8; ++vg) {
        const int row = rt * 16 + vg + 8 * hi;
        const int qi = row >> 4;
        const float u = (qRow[qi * CC + col] - (acc[vg] + bias)
                         + (float)PEbf[row * CC + col]) * scl;
        Ubf[row * CC + col] = (__bf16)u;
      }
    }
  }
  __syncthreads();

  // ---- gate layer 1: hidden = relu(u @ Wg1 + bg1) -> Abf (nb dead) ----
  {
    v16bf Breg[4];
    load_Btile(Wg1P, wv, lane, Breg);
    const int col = wv * 16 + m;
    const float bias = bg1[col];
    for (int rt = 0; rt < RR / 16; ++rt) {
      v8f acc = wmma_rowtile(Ubf, rt * 16, Breg, lane);
#pragma unroll
      for (int vg = 0; vg < 8; ++vg) {
        const int row = rt * 16 + vg + 8 * hi;
        Abf[row * CC + col] = (__bf16)fmaxf(acc[vg] + bias, 0.0f);
      }
    }
  }
  __syncthreads();

  // ---- gate layer 2 + softmax over K + weighted sum -> out ----
  // wave wv owns query qi = wv (its row tile covers all 16 neighbors).
  // D-tile column c: rows 0-7 in lane c, rows 8-15 in lane c+16 -> xor-16 reduce.
  {
    const int qi = wv;
    for (int ct = 0; ct < 8; ++ct) {
      v16bf Breg[4];
      load_Btile(Wg2P, ct, lane, Breg);
      v8f acc = wmma_rowtile(Abf, qi * 16, Breg, lane);
      const int col = ct * 16 + m;
      const float bias = bg2[col];
      float mx = -3.4e38f;
#pragma unroll
      for (int vg = 0; vg < 8; ++vg) {
        acc[vg] += bias;
        mx = fmaxf(mx, acc[vg]);
      }
      mx = fmaxf(mx, __shfl_xor(mx, 16, 32));
      float s = 0.0f, o = 0.0f;
#pragma unroll
      for (int vg = 0; vg < 8; ++vg) {
        const float e = __expf(acc[vg] - mx);
        const int row = qi * 16 + vg + 8 * hi;
        const float w = (float)Vbf[row * CC + col] + (float)PEbf[row * CC + col];
        s += e;
        o = fmaf(e, w, o);
      }
      s += __shfl_xor(s, 16, 32);
      o += __shfl_xor(o, 16, 32);
      o /= s;
      if (hi == 0) out[((size_t)b * NN + n0 + qi) * CC + col] = o;
    }
  }
}

// ---------------------------------------------------------------------------
// Launch
// ---------------------------------------------------------------------------
extern "C" void kernel_launch(void* const* d_in, const int* in_sizes, int n_in,
                              void* d_out, int out_size, void* d_ws, size_t ws_size,
                              hipStream_t stream) {
  (void)in_sizes; (void)n_in; (void)out_size; (void)ws_size;

  const float* x    = (const float*)d_in[0];
  const float* pos  = (const float*)d_in[1];
  const float* Wq   = (const float*)d_in[2];
  const float* bq   = (const float*)d_in[3];
  const float* Wk   = (const float*)d_in[4];
  const float* bk   = (const float*)d_in[5];
  const float* Wv   = (const float*)d_in[6];
  const float* bv   = (const float*)d_in[7];
  const float* Wp1  = (const float*)d_in[8];
  const float* bp1  = (const float*)d_in[9];
  const float* Wp2  = (const float*)d_in[10];
  const float* bp2  = (const float*)d_in[11];
  const float* Wg1  = (const float*)d_in[12];
  const float* bg1  = (const float*)d_in[13];
  const float* Wg2  = (const float*)d_in[14];
  const float* bg2  = (const float*)d_in[15];
  float* outp = (float*)d_out;

  // workspace: [knn idx: B*N*K ints][6 packed bf16 weights of 128x128]
  int* knnIdx = (int*)d_ws;
  __bf16* packBase = (__bf16*)((char*)d_ws + (size_t)BB * NN * KK * sizeof(int));
  __bf16* WqP  = packBase + 0 * (size_t)CC * CC;
  __bf16* WkP  = packBase + 1 * (size_t)CC * CC;
  __bf16* WvP  = packBase + 2 * (size_t)CC * CC;
  __bf16* Wp2P = packBase + 3 * (size_t)CC * CC;
  __bf16* Wg1P = packBase + 4 * (size_t)CC * CC;
  __bf16* Wg2P = packBase + 5 * (size_t)CC * CC;

  knn_kernel<<<(BB * NN) / 256, 256, 0, stream>>>(pos, knnIdx);

  pack_weight<<<32, 32, 0, stream>>>(Wq,  WqP);
  pack_weight<<<32, 32, 0, stream>>>(Wk,  WkP);
  pack_weight<<<32, 32, 0, stream>>>(Wv,  WvP);
  pack_weight<<<32, 32, 0, stream>>>(Wp2, Wp2P);
  pack_weight<<<32, 32, 0, stream>>>(Wg1, Wg1P);
  pack_weight<<<32, 32, 0, stream>>>(Wg2, Wg2P);

  // dynamic LDS: 4 big bf16 bufs + qA + fp32 scratch = 143360 bytes (< 320 KB/WGP)
  const size_t smemBytes =
      (size_t)(4 * RR + 16) * CC * sizeof(__bf16) +
      (size_t)(QT * CC + RR * 4 + 3 * CC + CC) * sizeof(float);

  pt_fused<<<(BB * NN) / QT, 256, smemBytes, stream>>>(
      x, pos, knnIdx, WqP, WkP, WvP, Wp2P, Wg1P, Wg2P,
      bq, bk, bv, Wp1, bp1, bp2, bg1, bg2, outp);
}
